// Attention_13597866459683
// MI455X (gfx1250) — matrix operand order repaired and checked by reading, not compiled, not run on hardware
//
#include <hip/hip_runtime.h>

// ---------------------------------------------------------------------------
// MHA for MI455X (gfx1250, wave32, WMMA). bf16 operands, f32 accumulation.
//   D_MODEL=1024, H=16, d=64, b=2, s=2048, SCALE=8.
// Workspace layout (bytes), total ~48 MB:
//   Xb     @ 0         : 4096*1024 bf16  (8 MB)
//   Wqkvb  @ 8388608   : 3072*1024 bf16  (6 MB)  (rows permuted: Q|K|V blocks)
//   Woutb  @ 14680064  : 1024*1024 bf16  (2 MB)
//   Qb     @ 16777216  : [b,h,s,d] bf16  (8 MB)
//   Kb     @ 25165824  : [b,h,s,d] bf16  (8 MB)
//   Vtb    @ 33554432  : [b,h,d,s] bf16  (8 MB)  (transposed for B-fragments)
//   Ab     @ 41943040  : [b*s,1024] bf16 (8 MB)  (attention output)
// ---------------------------------------------------------------------------

typedef __attribute__((ext_vector_type(16))) __bf16 v16bf;
typedef __attribute__((ext_vector_type(8)))  __bf16 v8bf;
typedef __attribute__((ext_vector_type(8)))  float  v8f;

#define D_MODEL 1024
#define SEQ     2048
#define HEADS   16
#define DHEAD   64

__device__ __forceinline__ v8f wmma_bf16(v16bf a, v16bf b, v8f c) {
  return __builtin_amdgcn_wmma_f32_16x16x32_bf16(
      /*neg_a=*/false, a, /*neg_b=*/false, b,
      /*c_mod=*/(short)0, c, /*reuse_a=*/false, /*reuse_b=*/false);
}

// A-fragment (16x32 bf16, M x K): lane L<16 -> row row0+L, K {k0..k0+7, k0+16..k0+23}
//                                 lane L>=16 -> row row0+(L-16), K {k0+8.., k0+24..}
__device__ __forceinline__ v16bf load_a_frag(const __bf16* base, int ld,
                                             int row0, int k0, int lane) {
  const __bf16* p = base + (long)(row0 + (lane & 15)) * ld + k0 + ((lane >> 4) << 3);
  v8bf lo = *(const v8bf*)p;
  v8bf hi = *(const v8bf*)(p + 16);
  return __builtin_shufflevector(lo, hi, 0,1,2,3,4,5,6,7,8,9,10,11,12,13,14,15);
}

// B-fragment (32x16 bf16, K x N): lane L holds column col0+(L&15),
// K {k0+(L>>4)*8..+7, k0+16+(L>>4)*8..+7}. base is row-major [N][K] (i.e. B^T),
// so a column of B is a contiguous row of base.
__device__ __forceinline__ v16bf load_b_frag(const __bf16* base, int ld,
                                             int col0, int k0, int lane) {
  const __bf16* p = base + (long)(col0 + (lane & 15)) * ld + k0 + ((lane >> 4) << 3);
  v8bf lo = *(const v8bf*)p;
  v8bf hi = *(const v8bf*)(p + 16);
  return __builtin_shufflevector(lo, hi, 0,1,2,3,4,5,6,7,8,9,10,11,12,13,14,15);
}

// ---- CDNA5 async global->LDS copy (ASYNCcnt-tracked, 16B per lane) --------
__device__ __forceinline__ void async_copy_b128(unsigned lds_off,
                                                const __bf16* gptr) {
  unsigned long long g = (unsigned long long)gptr;
  asm volatile("global_load_async_to_lds_b128 %0, %1, off"
               :: "v"(lds_off), "v"(g) : "memory");
}
__device__ __forceinline__ void wait_async0() {
  asm volatile("s_wait_asynccnt 0" ::: "memory");
}
// Low 32 bits of a flat LDS-aperture pointer are the LDS byte address.
__device__ __forceinline__ unsigned lds_off_of(const void* p) {
  return (unsigned)(unsigned long long)p;
}

// ------------------------- precision conversion ----------------------------

__global__ void cvt_bf16_kernel(const float* __restrict__ in,
                                __bf16* __restrict__ out, int n) {
  for (int i = blockIdx.x * blockDim.x + threadIdx.x; i < n;
       i += gridDim.x * blockDim.x)
    out[i] = (__bf16)in[i];
}

// W_qkv row r corresponds to (h, dq, c) with r = h*192 + dq*3 + c  (c: q/k/v).
// Permute so output row (c*1024 + h*64 + dq) = source row (h*192 + dq*3 + c).
__global__ void cvt_wqkv_kernel(const float* __restrict__ W,
                                __bf16* __restrict__ Wb) {
  const int n = 3 * D_MODEL * D_MODEL;
  for (int i = blockIdx.x * blockDim.x + threadIdx.x; i < n;
       i += gridDim.x * blockDim.x) {
    int c   = i >> 20;            // /(1024*1024)
    int rem = i & 0xFFFFF;
    int row = rem >> 10;          // h*64+dq
    int col = rem & 1023;
    int h = row >> 6, dq = row & 63;
    int src = (h * 192 + dq * 3 + c) * D_MODEL + col;
    Wb[i] = (__bf16)W[src];
  }
}

// ------------------------------ QKV GEMM -----------------------------------
// Y[4096 x 3072] = Xb @ Wqkvb^T ; scatter into Q/K/V (V transposed).
// Block = 8 waves, each wave does a 64x64 tile; block tile = 128(M) x 256(N).

__global__ __launch_bounds__(256) void qkv_gemm_kernel(
    const __bf16* __restrict__ X, const __bf16* __restrict__ W,
    __bf16* __restrict__ Qb, __bf16* __restrict__ Kb, __bf16* __restrict__ Vtb) {
  const int lane = threadIdx.x & 31;
  const int wave = threadIdx.x >> 5;
  const int m0 = blockIdx.x * 128 + (wave >> 2) * 64;
  const int n0 = blockIdx.y * 256 + (wave & 3) * 64;

  v8f acc[4][4] = {};
  for (int k0 = 0; k0 < D_MODEL; k0 += 32) {
    v16bf a[4], b[4];
#pragma unroll
    for (int i = 0; i < 4; ++i) a[i] = load_a_frag(X, D_MODEL, m0 + 16 * i, k0, lane);
#pragma unroll
    for (int j = 0; j < 4; ++j) b[j] = load_b_frag(W, D_MODEL, n0 + 16 * j, k0, lane);
#pragma unroll
    for (int i = 0; i < 4; ++i)
#pragma unroll
      for (int j = 0; j < 4; ++j)
        acc[i][j] = wmma_bf16(a[i], b[j], acc[i][j]);
  }

  const int hi = lane >> 4;
  const int ln = lane & 15;
#pragma unroll
  for (int i = 0; i < 4; ++i)
#pragma unroll
    for (int j = 0; j < 4; ++j) {
      int ncol = n0 + 16 * j + ln;     // 0..3071
      int c    = ncol >> 10;           // 0=Q 1=K 2=V
      int rem  = ncol & 1023;
      int h = rem >> 6, dq = rem & 63;
#pragma unroll
      for (int r = 0; r < 8; ++r) {
        int m  = m0 + 16 * i + r + 8 * hi;   // 0..4095
        int bb = m >> 11, ss = m & (SEQ - 1);
        long bh = (long)(bb * HEADS + h);
        __bf16 v = (__bf16)acc[i][j][r];
        if (c == 0)      Qb[(bh * SEQ + ss) * DHEAD + dq] = v;
        else if (c == 1) Kb[(bh * SEQ + ss) * DHEAD + dq] = v;
        else             Vtb[(bh * DHEAD + dq) * SEQ + ss] = v;
      }
    }
}

// --------------------------- flash attention -------------------------------
// Block = 8 waves = 128 queries of one (b,h). K/V tiles (64 keys) are staged
// into LDS once per block via async global->LDS copies (double-buffered),
// shared by all 8 waves -> 8x less L2 traffic than per-wave streaming.

#define KT_LDP 72   // padded row pitch (bf16 elems): 144B -> conflict-free

__global__ __launch_bounds__(256) void attn_kernel(
    const __bf16* __restrict__ Qb, const __bf16* __restrict__ Kb,
    const __bf16* __restrict__ Vtb, const unsigned char* __restrict__ mask,
    __bf16* __restrict__ O) {
  __shared__ __bf16 KtS[2][DHEAD][KT_LDP];   // [buf][key][d]
  __shared__ __bf16 VtS[2][DHEAD][KT_LDP];   // [buf][d][key]
  __shared__ __bf16 Plds[8][16][KT_LDP];     // per-wave P tile

  const int tid  = threadIdx.x;
  const int lane = tid & 31;
  const int wave = tid >> 5;
  const int hi = lane >> 4, ln = lane & 15;
  const int h = blockIdx.y, b = blockIdx.z;
  const int bh = b * HEADS + h;
  const int m0 = blockIdx.x * 128 + wave * 16;   // query row tile in [0,2048)

  const __bf16* Q  = Qb  + (long)bh * SEQ * DHEAD;
  const __bf16* K  = Kb  + (long)bh * SEQ * DHEAD;
  const __bf16* Vt = Vtb + (long)bh * DHEAD * SEQ;
  const unsigned char* mrow = mask + b * SEQ;

  const float C = 0.18033688011112042f;  // log2(e) / sqrt(64)

  v16bf qa0 = load_a_frag(Q, DHEAD, m0, 0,  lane);
  v16bf qa1 = load_a_frag(Q, DHEAD, m0, 32, lane);

  v8f acc[4] = {};                 // 16 x 64 output accumulator (f32)
  float mrun[8], lrun[8];
#pragma unroll
  for (int r = 0; r < 8; ++r) { mrun[r] = -3.0e38f; lrun[r] = 0.f; }

  // Stage one 64-key tile (K: [key][d], Vt: [d][key]) into buffer `buf`.
  // 512 16B-chunks per matrix, 2 per thread per matrix.
  auto stage = [&](int buf, int n0) {
#pragma unroll
    for (int q = 0; q < 2; ++q) {
      int c   = tid + 256 * q;
      int row = c >> 3;
      int col = (c & 7) << 3;
      async_copy_b128(lds_off_of(&KtS[buf][row][col]),
                      K + (long)(n0 + row) * DHEAD + col);
      async_copy_b128(lds_off_of(&VtS[buf][row][col]),
                      Vt + (long)row * SEQ + n0 + col);
    }
  };

  stage(0, 0);
  wait_async0();
  __syncthreads();

  const int NT = SEQ / 64;   // 32 key tiles
  for (int it = 0; it < NT; ++it) {
    const int cur = it & 1;
    const int n0  = it * 64;
    if (it + 1 < NT) stage(1 - cur, n0 + 64);

    const __bf16* KtL = &KtS[cur][0][0];
    const __bf16* VtL = &VtS[cur][0][0];

    // ---- S = Q @ K^T for 4 key sub-tiles (B-frags from LDS) ------------
    v8f s[4];
#pragma unroll
    for (int j = 0; j < 4; ++j) {
      v16bf kb0 = load_b_frag(KtL, KT_LDP, 16 * j, 0,  lane);
      v16bf kb1 = load_b_frag(KtL, KT_LDP, 16 * j, 32, lane);
      v8f t = {};
      t = wmma_bf16(qa0, kb0, t);
      t = wmma_bf16(qa1, kb1, t);
      s[j] = t;
    }
    // mask bias (column fixed per lane per sub-tile)
#pragma unroll
    for (int j = 0; j < 4; ++j) {
      float bias = mrow[n0 + 16 * j + ln] ? 0.f : -3.0e38f;
#pragma unroll
      for (int r = 0; r < 8; ++r) s[j][r] += bias;
    }
    // ---- online softmax: row max / sum across 16 lanes of half-wave ----
    float mnew[8], corr[8];
#pragma unroll
    for (int r = 0; r < 8; ++r) {
      float v = fmaxf(fmaxf(s[0][r], s[1][r]), fmaxf(s[2][r], s[3][r]));
      v = fmaxf(v, __shfl_xor(v, 1, 32));
      v = fmaxf(v, __shfl_xor(v, 2, 32));
      v = fmaxf(v, __shfl_xor(v, 4, 32));
      v = fmaxf(v, __shfl_xor(v, 8, 32));
      mnew[r] = fmaxf(mrun[r], v);
      corr[r] = __builtin_amdgcn_exp2f((mrun[r] - mnew[r]) * C);
    }
#pragma unroll
    for (int r = 0; r < 8; ++r) {
      float rs = 0.f;
#pragma unroll
      for (int j = 0; j < 4; ++j) {
        float p = __builtin_amdgcn_exp2f((s[j][r] - mnew[r]) * C);
        s[j][r] = p;
        rs += p;
      }
      rs += __shfl_xor(rs, 1, 32);
      rs += __shfl_xor(rs, 2, 32);
      rs += __shfl_xor(rs, 4, 32);
      rs += __shfl_xor(rs, 8, 32);
      lrun[r] = lrun[r] * corr[r] + rs;
      mrun[r] = mnew[r];
    }
#pragma unroll
    for (int t = 0; t < 4; ++t)
#pragma unroll
      for (int r = 0; r < 8; ++r) acc[t][r] *= corr[r];

    // ---- C-layout P -> LDS row-major -> A-fragment ---------------------
#pragma unroll
    for (int j = 0; j < 4; ++j)
#pragma unroll
      for (int r = 0; r < 8; ++r)
        Plds[wave][r + 8 * hi][16 * j + ln] = (__bf16)s[j][r];
    asm volatile("s_wait_dscnt 0" ::: "memory");  // same-wave DS in order

    const __bf16* Pp = &Plds[wave][0][0];
    v16bf pa0 = load_a_frag(Pp, KT_LDP, 0, 0,  lane);
    v16bf pa1 = load_a_frag(Pp, KT_LDP, 0, 32, lane);

    // ---- acc += P @ V (B-frags from LDS transposed-V tile) -------------
#pragma unroll
    for (int t = 0; t < 4; ++t) {
      v16bf vb0 = load_b_frag(VtL, KT_LDP, 16 * t, 0,  lane);
      v16bf vb1 = load_b_frag(VtL, KT_LDP, 16 * t, 32, lane);
      acc[t] = wmma_bf16(pa0, vb0, acc[t]);
      acc[t] = wmma_bf16(pa1, vb1, acc[t]);
    }

    wait_async0();      // next tile landed in LDS
    __syncthreads();    // all waves done with `cur` before it's overwritten
  }

  // ---- normalize and store to Ab[b*s][h*64+d] as bf16 -------------------
#pragma unroll
  for (int r = 0; r < 8; ++r) {
    float inv = 1.0f / lrun[r];
    int m = m0 + r + 8 * hi;
#pragma unroll
    for (int t = 0; t < 4; ++t) {
      float o = acc[t][r] * inv;
      O[((long)(b * SEQ + m)) * D_MODEL + h * DHEAD + 16 * t + ln] = (__bf16)o;
    }
  }
}

// ------------------------- output projection GEMM --------------------------
// out[4096 x 1024] (f32) = Ab @ Woutb^T

__global__ __launch_bounds__(256) void out_gemm_kernel(
    const __bf16* __restrict__ A, const __bf16* __restrict__ W,
    float* __restrict__ out) {
  const int lane = threadIdx.x & 31;
  const int wave = threadIdx.x >> 5;
  const int m0 = blockIdx.x * 128 + (wave >> 2) * 64;
  const int n0 = blockIdx.y * 256 + (wave & 3) * 64;

  v8f acc[4][4] = {};
  for (int k0 = 0; k0 < D_MODEL; k0 += 32) {
    v16bf a[4], b[4];
#pragma unroll
    for (int i = 0; i < 4; ++i) a[i] = load_a_frag(A, D_MODEL, m0 + 16 * i, k0, lane);
#pragma unroll
    for (int j = 0; j < 4; ++j) b[j] = load_b_frag(W, D_MODEL, n0 + 16 * j, k0, lane);
#pragma unroll
    for (int i = 0; i < 4; ++i)
#pragma unroll
      for (int j = 0; j < 4; ++j)
        acc[i][j] = wmma_bf16(a[i], b[j], acc[i][j]);
  }

  const int hi = lane >> 4, ln = lane & 15;
#pragma unroll
  for (int i = 0; i < 4; ++i)
#pragma unroll
    for (int j = 0; j < 4; ++j)
#pragma unroll
      for (int r = 0; r < 8; ++r) {
        int m = m0 + 16 * i + r + 8 * hi;
        out[(long)m * D_MODEL + n0 + 16 * j + ln] = acc[i][j][r];
      }
}

// ------------------------------- launch ------------------------------------

extern "C" void kernel_launch(void* const* d_in, const int* in_sizes, int n_in,
                              void* d_out, int out_size, void* d_ws, size_t ws_size,
                              hipStream_t stream) {
  const float* X     = (const float*)d_in[0];
  const float* Wqkv  = (const float*)d_in[1];
  const float* Wout  = (const float*)d_in[2];
  const unsigned char* mask = (const unsigned char*)d_in[3];
  float* out = (float*)d_out;

  char* ws = (char*)d_ws;
  __bf16* Xb    = (__bf16*)(ws + 0);
  __bf16* Wqkvb = (__bf16*)(ws + 8388608);
  __bf16* Woutb = (__bf16*)(ws + 14680064);
  __bf16* Qb    = (__bf16*)(ws + 16777216);
  __bf16* Kb    = (__bf16*)(ws + 25165824);
  __bf16* Vtb   = (__bf16*)(ws + 33554432);
  __bf16* Ab    = (__bf16*)(ws + 41943040);

  cvt_bf16_kernel<<<4096, 256, 0, stream>>>(X, Xb, 4096 * D_MODEL);
  cvt_wqkv_kernel<<<4096, 256, 0, stream>>>(Wqkv, Wqkvb);
  cvt_bf16_kernel<<<2048, 256, 0, stream>>>(Wout, Woutb, D_MODEL * D_MODEL);

  qkv_gemm_kernel<<<dim3(32, 12), 256, 0, stream>>>(Xb, Wqkvb, Qb, Kb, Vtb);
  attn_kernel<<<dim3(SEQ / 128, HEADS, 2), 256, 0, stream>>>(Qb, Kb, Vtb, mask, Ab);
  out_gemm_kernel<<<dim3(32, 4), 256, 0, stream>>>(Ab, Woutb, out);
}
